// TreeTransformer_35734127903540
// MI455X (gfx1250) — matrix operand order repaired and checked by reading, not compiled, not run on hardware
//
#include <hip/hip_runtime.h>

// ---------------- constants ----------------
#define BATCH   32
#define AGENTS  64
#define SEQS    (BATCH*AGENTS)   // 2048
#define N_TOK   121
#define NP      128              // padded tokens
#define FIN     12
#define H       128
#define HEADS   4
#define HD      32               // head dim
#define NLAYER  2
#define QKV_N   384
#define OUTF    256
#define KTOT    (NP*H)           // 16384 padded K for out-proj
#define EPS     1e-5f

typedef __bf16 bf16;
typedef __attribute__((ext_vector_type(16))) __bf16 v16bf;
typedef __attribute__((ext_vector_type(8)))  float  v8f;

// ---------------- WMMA helpers (wave32, 16x16x32 bf16 -> f32) ----------------
__device__ __forceinline__ v8f wmma_bf16(v16bf a, v16bf b, v8f c) {
  return __builtin_amdgcn_wmma_f32_16x16x32_bf16(false, a, false, b, (short)0, c,
                                                 false, false);
}

// A fragment: 16x32 (MxK) bf16 from row-major [m][k] storage (stride in elems).
// lane m = lane%16; khalf = lane/16; elem j -> K = kt*32 + khalf*16 + j
__device__ __forceinline__ v16bf load_a_frag(const bf16* base, int stride,
                                             int mtile, int kt, int lane) {
  int m = mtile * 16 + (lane & 15);
  int khalf = lane >> 4;
  const bf16* p = base + (size_t)m * stride + kt * 32 + khalf * 16;
  v16bf f;
#pragma unroll
  for (int j = 0; j < 16; j++) f[j] = p[j];
  return f;
}

// B fragment: 32x16 (KxN) where element (k,n) is stored at w[n*stride + k]
// (weight row-major [n][k]).  lane n = lane%16; elem j -> K = kt*32 + (lane/16)*16 + j
__device__ __forceinline__ v16bf load_b_frag_nk(const bf16* w, int stride,
                                                int ntile, int kt, int lane) {
  int n = ntile * 16 + (lane & 15);
  const bf16* p = w + (size_t)n * stride + kt * 32 + (lane >> 4) * 16;
  v16bf f;
#pragma unroll
  for (int j = 0; j < 16; j++) f[j] = p[j];
  return f;
}

// B fragment: 32x16 (KxN) where element (k,n) is stored at v[k*stride + n]
__device__ __forceinline__ v16bf load_b_frag_kn(const bf16* v, int stride,
                                                int ntile, int kt, int lane) {
  int n = ntile * 16 + (lane & 15);
  int k0 = kt * 32 + (lane >> 4) * 16;
  v16bf f;
#pragma unroll
  for (int j = 0; j < 16; j++) f[j] = v[(size_t)(k0 + j) * stride + n];
  return f;
}

// ---------------- weight conversion kernels ----------------
__global__ void cvt_f32_bf16_kernel(const float* __restrict__ src,
                                    bf16* __restrict__ dst, int n) {
  int i = blockIdx.x * blockDim.x + threadIdx.x;
  if (i < n) dst[i] = (bf16)src[i];
}

// pad w_out (OUTF x N_TOK*H) -> (OUTF x NP*H) bf16, zero in padded K
__global__ void pad_wout_kernel(const float* __restrict__ src,
                                bf16* __restrict__ dst) {
  int i = blockIdx.x * blockDim.x + threadIdx.x;
  if (i < OUTF * KTOT) {
    int o = i >> 14;        // / 16384
    int k = i & (KTOT - 1);
    int n = k >> 7;
    int h = k & 127;
    dst[i] = (n < N_TOK) ? (bf16)src[(size_t)o * (N_TOK * H) + n * H + h]
                         : (bf16)0.0f;
  }
}

// ---------------- fused per-sequence transformer kernel ----------------
// one block per sequence; 256 threads = 8 wave32
// dynamic LDS layout (bytes):
//  [0      ) float xres[128][128]   65536
//  [65536  ) bf16  xb  [128][128]   32768   (A operand staging / probs / ff h)
//  [98304  ) bf16  qkv [128][384]   98304
//  [196608 ) float sc  [128][128]   65536   (scores / pre-LN accum)
//  [262144 ) bf16  ao  [128][128]   32768   (attention out / ff1 out)
//  total 294912
#define SEQ_LDS_BYTES 294912

__global__ __launch_bounds__(256)
void tree_seq_kernel(const float* __restrict__ forest,
                     const int*   __restrict__ perm,
                     const float* __restrict__ w_in,
                     const float* __restrict__ b_in,
                     const bf16*  __restrict__ qkvw,
                     const float* __restrict__ qkvb,
                     const bf16*  __restrict__ projw,
                     const float* __restrict__ projb,
                     const bf16*  __restrict__ ff1w,
                     const float* __restrict__ ff1b,
                     const bf16*  __restrict__ ff2w,
                     const float* __restrict__ ff2b,
                     const float* __restrict__ ln1g,
                     const float* __restrict__ ln1b,
                     const float* __restrict__ ln2g,
                     const float* __restrict__ ln2b,
                     bf16* __restrict__ Xout) {
  extern __shared__ char smem[];
  float* xres = (float*)(smem);
  bf16*  xb   = (bf16*)(smem + 65536);
  bf16*  qkv  = (bf16*)(smem + 98304);
  float* sc   = (float*)(smem + 196608);
  bf16*  ao   = (bf16*)(smem + 262144);

  const int s    = blockIdx.x;
  const int tid  = threadIdx.x;
  const int lane = tid & 31;
  const int wave = tid >> 5;

  // ---- embed: x = forest @ w_in.T + b_in + tree_pe, permuted tokens ----
  if (tid < NP) {
    int i = tid;
    if (i < N_TOK) {
      int node = perm[i];
      float f[FIN];
      const float* fp = forest + ((size_t)s * N_TOK + node) * FIN;
#pragma unroll
      for (int j = 0; j < FIN; j++) f[j] = fp[j];
      // tree positional encoding: bit (depth-1)*3 + branch along root path
      unsigned mask = 0;
      {
        int depth = 0, t2 = node;
        while (t2 > 0) { t2 = (t2 - 1) / 3; depth++; }
        int c = node, dd = depth;
        while (c > 0) {
          int br = (c - 1) % 3;
          mask |= 1u << ((dd - 1) * 3 + br);
          c = (c - 1) / 3;
          dd--;
        }
      }
      for (int h = 0; h < H; h++) {
        const float* w = w_in + h * FIN;
        float acc = b_in[h];
#pragma unroll
        for (int j = 0; j < FIN; j++) acc += f[j] * w[j];
        if (h < 12 && ((mask >> h) & 1u)) acc += 1.0f;
        xres[i * H + h] = acc;
        xb[i * H + h] = (bf16)acc;
      }
    } else {
      for (int h = 0; h < H; h++) { xres[i * H + h] = 0.0f; xb[i * H + h] = (bf16)0.0f; }
    }
  }
  __syncthreads();

  for (int L = 0; L < NLAYER; L++) {
    const bf16*  Wq = qkvw + (size_t)L * QKV_N * H;
    const float* Bq = qkvb + L * QKV_N;
    const bf16*  Wp = projw + (size_t)L * H * H;
    const float* Bp = projb + L * H;
    const bf16*  W1 = ff1w + (size_t)L * H * H;
    const float* B1 = ff1b + L * H;
    const bf16*  W2 = ff2w + (size_t)L * H * H;
    const float* B2 = ff2b + L * H;

    // ---- QKV: [128x128] x [128x384] ----
    for (int job = wave; job < 8 * 24; job += 8) {
      int mt = job / 24, nt = job % 24;
      v8f acc = {};
#pragma unroll
      for (int kt = 0; kt < 4; kt++) {
        v16bf a = load_a_frag(xb, H, mt, kt, lane);
        v16bf b = load_b_frag_nk(Wq, H, nt, kt, lane);
        acc = wmma_bf16(a, b, acc);
      }
      int n  = nt * 16 + (lane & 15);
      int mb = mt * 16 + ((lane >> 4) * 8);
      float bias = Bq[n];
#pragma unroll
      for (int r = 0; r < 8; r++)
        qkv[(size_t)(mb + r) * QKV_N + n] = (bf16)(acc[r] + bias);
    }
    __syncthreads();

    // ---- attention, head by head ----
    const float scale = 0.1767766952966369f;  // 1/sqrt(32)
    for (int hh = 0; hh < HEADS; hh++) {
      // scores = Q Kt : M=128, N=128, K=32 (one wmma per tile)
      for (int job = wave; job < 64; job += 8) {
        int mt = job >> 3, nt = job & 7;
        v8f acc = {};
        v16bf a = load_a_frag(qkv + hh * HD, QKV_N, mt, 0, lane);
        v16bf b = load_b_frag_nk(qkv + H + hh * HD, QKV_N, nt, 0, lane);
        acc = wmma_bf16(a, b, acc);
        int n  = nt * 16 + (lane & 15);
        int mb = mt * 16 + ((lane >> 4) * 8);
#pragma unroll
        for (int r = 0; r < 8; r++) sc[(mb + r) * NP + n] = acc[r] * scale;
      }
      __syncthreads();

      // row softmax (one row per thread), write probs bf16 into xb
      if (tid < NP) {
        int row = tid;
        if (row < N_TOK) {
          float mx = -1e30f;
          for (int j = 0; j < N_TOK; j++) mx = fmaxf(mx, sc[row * NP + j]);
          float sum = 0.0f;
          for (int j = 0; j < N_TOK; j++) {
            float e = __expf(sc[row * NP + j] - mx);
            sc[row * NP + j] = e;
            sum += e;
          }
          float inv = 1.0f / sum;
          for (int j = 0; j < N_TOK; j++) xb[row * NP + j] = (bf16)(sc[row * NP + j] * inv);
          for (int j = N_TOK; j < NP; j++) xb[row * NP + j] = (bf16)0.0f;
        } else {
          for (int j = 0; j < NP; j++) xb[row * NP + j] = (bf16)0.0f;
        }
      }
      __syncthreads();

      // out_h = P V : M=128, N=32, K=128
      for (int job = wave; job < 16; job += 8) {
        int mt = job >> 1, nt = job & 1;
        v8f acc = {};
#pragma unroll
        for (int kt = 0; kt < 4; kt++) {
          v16bf a = load_a_frag(xb, NP, mt, kt, lane);
          v16bf b = load_b_frag_kn(qkv + 2 * H + hh * HD, QKV_N, nt, kt, lane);
          acc = wmma_bf16(a, b, acc);
        }
        int n  = nt * 16 + (lane & 15);
        int mb = mt * 16 + ((lane >> 4) * 8);
#pragma unroll
        for (int r = 0; r < 8; r++) ao[(mb + r) * H + hh * HD + n] = (bf16)acc[r];
      }
      __syncthreads();
    }

    // ---- proj + residual -> sc ----
    for (int job = wave; job < 64; job += 8) {
      int mt = job >> 3, nt = job & 7;
      v8f acc = {};
#pragma unroll
      for (int kt = 0; kt < 4; kt++) {
        v16bf a = load_a_frag(ao, H, mt, kt, lane);
        v16bf b = load_b_frag_nk(Wp, H, nt, kt, lane);
        acc = wmma_bf16(a, b, acc);
      }
      int n  = nt * 16 + (lane & 15);
      int mb = mt * 16 + ((lane >> 4) * 8);
      float bias = Bp[n];
#pragma unroll
      for (int r = 0; r < 8; r++) {
        int m = mb + r;
        sc[m * H + n] = acc[r] + bias + xres[m * H + n];
      }
    }
    __syncthreads();

    // ---- LN1 -> xres (f32) + xb (bf16) ----
    if (tid < NP) {
      int row = tid;
      if (row < N_TOK) {
        float m = 0.0f;
        for (int j = 0; j < H; j++) m += sc[row * H + j];
        m *= (1.0f / H);
        float v = 0.0f;
        for (int j = 0; j < H; j++) { float d = sc[row * H + j] - m; v += d * d; }
        float r = rsqrtf(v * (1.0f / H) + EPS);
        const float* g = ln1g + L * H;
        const float* b = ln1b + L * H;
        for (int j = 0; j < H; j++) {
          float val = (sc[row * H + j] - m) * r * g[j] + b[j];
          xres[row * H + j] = val;
          xb[row * H + j] = (bf16)val;
        }
      } else {
        for (int j = 0; j < H; j++) { xres[row * H + j] = 0.0f; xb[row * H + j] = (bf16)0.0f; }
      }
    }
    __syncthreads();

    // ---- FF1 + ReLU -> ao (bf16) ----
    for (int job = wave; job < 64; job += 8) {
      int mt = job >> 3, nt = job & 7;
      v8f acc = {};
#pragma unroll
      for (int kt = 0; kt < 4; kt++) {
        v16bf a = load_a_frag(xb, H, mt, kt, lane);
        v16bf b = load_b_frag_nk(W1, H, nt, kt, lane);
        acc = wmma_bf16(a, b, acc);
      }
      int n  = nt * 16 + (lane & 15);
      int mb = mt * 16 + ((lane >> 4) * 8);
      float bias = B1[n];
#pragma unroll
      for (int r = 0; r < 8; r++)
        ao[(mb + r) * H + n] = (bf16)fmaxf(acc[r] + bias, 0.0f);
    }
    __syncthreads();

    // ---- FF2 + residual -> sc ----
    for (int job = wave; job < 64; job += 8) {
      int mt = job >> 3, nt = job & 7;
      v8f acc = {};
#pragma unroll
      for (int kt = 0; kt < 4; kt++) {
        v16bf a = load_a_frag(ao, H, mt, kt, lane);
        v16bf b = load_b_frag_nk(W2, H, nt, kt, lane);
        acc = wmma_bf16(a, b, acc);
      }
      int n  = nt * 16 + (lane & 15);
      int mb = mt * 16 + ((lane >> 4) * 8);
      float bias = B2[n];
#pragma unroll
      for (int r = 0; r < 8; r++) {
        int m = mb + r;
        sc[m * H + n] = acc[r] + bias + xres[m * H + n];
      }
    }
    __syncthreads();

    // ---- LN2 -> xres + xb ----
    if (tid < NP) {
      int row = tid;
      if (row < N_TOK) {
        float m = 0.0f;
        for (int j = 0; j < H; j++) m += sc[row * H + j];
        m *= (1.0f / H);
        float v = 0.0f;
        for (int j = 0; j < H; j++) { float d = sc[row * H + j] - m; v += d * d; }
        float r = rsqrtf(v * (1.0f / H) + EPS);
        const float* g = ln2g + L * H;
        const float* b = ln2b + L * H;
        for (int j = 0; j < H; j++) {
          float val = (sc[row * H + j] - m) * r * g[j] + b[j];
          xres[row * H + j] = val;
          xb[row * H + j] = (bf16)val;
        }
      } else {
        for (int j = 0; j < H; j++) { xres[row * H + j] = 0.0f; xb[row * H + j] = (bf16)0.0f; }
      }
    }
    __syncthreads();
  }

  // ---- write padded bf16 activations for output GEMM ----
  for (int idx = tid; idx < NP * H; idx += 256)
    Xout[(size_t)s * (NP * H) + idx] = xb[idx];
}

// ---------------- output projection + final LN ----------------
// grid = 128 blocks, each computes 16 rows x 256 cols, K = 16384
__global__ __launch_bounds__(256)
void outproj_kernel(const bf16* __restrict__ X,
                    const bf16* __restrict__ Wout,   // [256][16384] bf16, row-major [n][k]
                    const float* __restrict__ b_out,
                    const float* __restrict__ lnfg,
                    const float* __restrict__ lnfb,
                    float* __restrict__ out) {
  __shared__ float ytile[16][OUTF];
  __shared__ float stat[16][2];

  const int blk  = blockIdx.x;
  const int tid  = threadIdx.x;
  const int lane = tid & 31;
  const int wave = tid >> 5;

  const bf16* Arow = X + (size_t)blk * 16 * KTOT;
  const int nt0 = wave * 2;

  v8f acc0 = {};
  v8f acc1 = {};
#pragma unroll 2
  for (int kt = 0; kt < KTOT / 32; kt++) {
    // stream w_out through L2; prefetch the next K-slab
    __builtin_prefetch(Wout + (size_t)(nt0 * 16) * KTOT + (kt + 4) * 32, 0, 1);
    v16bf a  = load_a_frag(Arow, KTOT, 0, kt, lane);
    v16bf b0 = load_b_frag_nk(Wout, KTOT, nt0 + 0, kt, lane);
    v16bf b1 = load_b_frag_nk(Wout, KTOT, nt0 + 1, kt, lane);
    acc0 = wmma_bf16(a, b0, acc0);
    acc1 = wmma_bf16(a, b1, acc1);
  }
  {
    int mb = (lane >> 4) * 8;
    int nl = lane & 15;
#pragma unroll
    for (int r = 0; r < 8; r++) {
      int m  = mb + r;
      int n0 = (nt0 + 0) * 16 + nl;
      int n1 = (nt0 + 1) * 16 + nl;
      ytile[m][n0] = acc0[r] + b_out[n0];
      ytile[m][n1] = acc1[r] + b_out[n1];
    }
  }
  __syncthreads();

  if (tid < 16) {
    float m = 0.0f;
    for (int j = 0; j < OUTF; j++) m += ytile[tid][j];
    m *= (1.0f / OUTF);
    float v = 0.0f;
    for (int j = 0; j < OUTF; j++) { float d = ytile[tid][j] - m; v += d * d; }
    stat[tid][0] = m;
    stat[tid][1] = rsqrtf(v * (1.0f / OUTF) + EPS);
  }
  __syncthreads();

  for (int idx = tid; idx < 16 * OUTF; idx += 256) {
    int r = idx >> 8;
    int c = idx & (OUTF - 1);
    out[((size_t)blk * 16 + r) * OUTF + c] =
        (ytile[r][c] - stat[r][0]) * stat[r][1] * lnfg[c] + lnfb[c];
  }
}

// ---------------- host launcher ----------------
extern "C" void kernel_launch(void* const* d_in, const int* in_sizes, int n_in,
                              void* d_out, int out_size, void* d_ws, size_t ws_size,
                              hipStream_t stream) {
  (void)in_sizes; (void)n_in; (void)out_size; (void)ws_size;

  const float* forest = (const float*)d_in[0];
  // d_in[1] = adjacency (fixed ternary heap; PE derived analytically from node id)
  const int*   perm   = (const int*)d_in[2];
  const float* w_in   = (const float*)d_in[3];
  const float* b_in   = (const float*)d_in[4];
  const float* qkv_w  = (const float*)d_in[5];
  const float* qkv_b  = (const float*)d_in[6];
  const float* proj_w = (const float*)d_in[7];
  const float* proj_b = (const float*)d_in[8];
  const float* ff1_w  = (const float*)d_in[9];
  const float* ff1_b  = (const float*)d_in[10];
  const float* ff2_w  = (const float*)d_in[11];
  const float* ff2_b  = (const float*)d_in[12];
  const float* ln1_g  = (const float*)d_in[13];
  const float* ln1_b  = (const float*)d_in[14];
  const float* ln2_g  = (const float*)d_in[15];
  const float* ln2_b  = (const float*)d_in[16];
  const float* w_out  = (const float*)d_in[17];
  const float* b_out  = (const float*)d_in[18];
  const float* lnf_g  = (const float*)d_in[19];
  const float* lnf_b  = (const float*)d_in[20];
  float* out = (float*)d_out;

  // workspace layout
  char* ws = (char*)d_ws;
  size_t off = 0;
  bf16* Xb = (bf16*)(ws + off);           off += (size_t)SEQS * NP * H * 2;    // 64 MB
  bf16* qkvw_b = (bf16*)(ws + off);       off += (size_t)NLAYER * QKV_N * H * 2;
  bf16* projw_b = (bf16*)(ws + off);      off += (size_t)NLAYER * H * H * 2;
  bf16* ff1w_b = (bf16*)(ws + off);       off += (size_t)NLAYER * H * H * 2;
  bf16* ff2w_b = (bf16*)(ws + off);       off += (size_t)NLAYER * H * H * 2;
  bf16* woutp = (bf16*)(ws + off);        off += (size_t)OUTF * KTOT * 2;      // 8 MB

  // weight conversions (fp32 -> bf16, padded w_out)
  {
    int n;
    n = NLAYER * QKV_N * H;
    cvt_f32_bf16_kernel<<<(n + 255) / 256, 256, 0, stream>>>(qkv_w, qkvw_b, n);
    n = NLAYER * H * H;
    cvt_f32_bf16_kernel<<<(n + 255) / 256, 256, 0, stream>>>(proj_w, projw_b, n);
    cvt_f32_bf16_kernel<<<(n + 255) / 256, 256, 0, stream>>>(ff1_w, ff1w_b, n);
    cvt_f32_bf16_kernel<<<(n + 255) / 256, 256, 0, stream>>>(ff2_w, ff2w_b, n);
    n = OUTF * KTOT;
    pad_wout_kernel<<<(n + 255) / 256, 256, 0, stream>>>(w_out, woutp);
  }

  // fused per-sequence transformer (288 KB dynamic LDS per workgroup)
  tree_seq_kernel<<<SEQS, 256, SEQ_LDS_BYTES, stream>>>(
      forest, perm, w_in, b_in,
      qkvw_b, qkv_b, projw_b, proj_b,
      ff1w_b, ff1_b, ff2w_b, ff2_b,
      ln1_g, ln1_b, ln2_g, ln2_b, Xb);

  // output projection + final LN
  outproj_kernel<<<SEQS / 16, 256, 0, stream>>>(Xb, woutp, b_out, lnf_g, lnf_b, out);
}
